// CKGConv3d_31945966748001
// MI455X (gfx1250) — matrix operand order repaired and checked by reading, not compiled, not run on hardware
//
#include <hip/hip_runtime.h>
#include <hip/hip_bf16.h>

typedef __attribute__((ext_vector_type(16))) _Float16 v16h;
typedef __attribute__((ext_vector_type(8)))  _Float16 v8h;
typedef __attribute__((ext_vector_type(8)))  float    v8f;

#define BB    8
#define CIN   32
#define COUT  64
#define KK    4
#define DD    32
#define HH    64
#define WW    64
#define KTOT  (CIN * 27)        // 864, K ordered tap-major: k = t*32 + i
#define DHW   (DD * HH * WW)    // 131072
#define ASTRIDE 872             // padded LDS row stride in halfs (1744 B, 16B-aligned)

__device__ __forceinline__ float gelu_erf(float t) {
    return 0.5f * t * (1.0f + erff(t * 0.70710678118654752f));
}

// ---------------------------------------------------------------- pool: p[b,c] = mean(x[b,c,:,:,:])
__global__ __launch_bounds__(256)
void pool_kernel(const float* __restrict__ x, float* __restrict__ p) {
    __shared__ float red[256];
    const int bc = blockIdx.x;                       // b*CIN + c
    const float* base = x + (size_t)bc * DHW;
    float s = 0.f;
    for (int i = threadIdx.x; i < DHW; i += 256) s += base[i];
    red[threadIdx.x] = s;
    __syncthreads();
    for (int st = 128; st > 0; st >>= 1) {
        if (threadIdx.x < st) red[threadIdx.x] += red[threadIdx.x + st];
        __syncthreads();
    }
    if (threadIdx.x == 0) p[bc] = red[0] * (1.0f / DHW);
}

// ---------------------------------------------------------------- attention MLP + softmax (tiny)
__global__ void attn_kernel(const float* __restrict__ p,
                            const float* g1, const float* bb1,
                            const float* w1, const float* l1b,
                            const float* g2, const float* bb2,
                            const float* w2, const float* l2b,
                            const float* g3, const float* bb3,
                            const float* w3, const float* l3b,
                            const float* temp,
                            float* __restrict__ attn) {
    __shared__ float s0[CIN], s1[CIN];
    const int c = threadIdx.x;                       // 32 threads
    const float T = temp[0];
    for (int b = 0; b < BB; ++b) {
        float pc = p[b * CIN + c];
        float a1 = gelu_erf(pc * g1[c] + bb1[c]);
        s0[c] = a1;
        __syncthreads();
        float h1 = l1b[c];
        for (int i = 0; i < CIN; ++i) h1 += s0[i] * w1[c * CIN + i];
        float a2 = gelu_erf(h1 * g2[c] + bb2[c]);
        s1[c] = a2;
        __syncthreads();
        float h2 = l2b[c];
        for (int i = 0; i < CIN; ++i) h2 += s1[i] * w2[c * CIN + i];
        float a3 = (h2 + pc) * g3[c] + bb3[c];
        __syncthreads();
        s0[c] = a3;
        __syncthreads();
        if (c == 0) {
            float lg[KK], mx = -1e30f;
            for (int k = 0; k < KK; ++k) {
                float v = l3b[k];
                for (int i = 0; i < CIN; ++i) v += s0[i] * w3[k * CIN + i];
                v /= T;
                lg[k] = v;
                mx = fmaxf(mx, v);
            }
            float sum = 0.f;
            for (int k = 0; k < KK; ++k) { lg[k] = __expf(lg[k] - mx); sum += lg[k]; }
            for (int k = 0; k < KK; ++k) attn[b * KK + k] = lg[k] / sum;
        }
        __syncthreads();
    }
}

// ---------------------------------------------------------------- aggregate kernels -> f16 (tap-major K)
__global__ __launch_bounds__(256)
void aggw_kernel(const float* __restrict__ weight, const float* __restrict__ cbias,
                 const float* __restrict__ attn,
                 _Float16* __restrict__ aggW, float* __restrict__ aggB) {
    const int bo = blockIdx.x;                       // b*COUT + o
    const int b = bo >> 6, o = bo & 63;
    const float a0 = attn[b * KK + 0], a1 = attn[b * KK + 1];
    const float a2 = attn[b * KK + 2], a3 = attn[b * KK + 3];
    const float* w0 = weight + (size_t)(0 * COUT + o) * KTOT;
    const float* w1 = weight + (size_t)(1 * COUT + o) * KTOT;
    const float* w2 = weight + (size_t)(2 * COUT + o) * KTOT;
    const float* w3 = weight + (size_t)(3 * COUT + o) * KTOT;
    _Float16* dst = aggW + (size_t)bo * KTOT;
    for (int idx = threadIdx.x; idx < KTOT; idx += 256) {
        const int t = idx >> 5, i = idx & 31;        // dst k = t*32 + i
        const int s = i * 27 + t;                    // weight inner layout [i][t]
        float v = a0 * w0[s] + a1 * w1[s] + a2 * w2[s] + a3 * w3[s];
        dst[idx] = (_Float16)v;
    }
    if (threadIdx.x == 0)
        aggB[bo] = a0 * cbias[0 * COUT + o] + a1 * cbias[1 * COUT + o]
                 + a2 * cbias[2 * COUT + o] + a3 * cbias[3 * COUT + o];
}

// ---------------------------------------------------------------- implicit-GEMM conv3d via WMMA
// grid: (512 n-groups, 8 batch); block = 256 = 8 wave32.
// All 64 weight rows staged in LDS (111.6 KB of the 320 KB WGP pool).
// K-step = one conv tap x 32 input channels. Each wave: TWO 16-position
// N-tiles (a 32-wide half row) x FOUR 16-channel M-tiles = 8 WMMAs/tap.
// Each A fragment (2x ds_load_b128) feeds 2 WMMAs; each B gather feeds 4.
__global__ __launch_bounds__(256)
void conv_wmma_kernel(const float* __restrict__ x,
                      const _Float16* __restrict__ aggW,
                      const float* __restrict__ aggB,
                      float* __restrict__ out) {
    __shared__ _Float16 sA[COUT * ASTRIDE];          // 111,616 B: all 64 weight rows

    const int b = blockIdx.y;
    // ---- cooperative A load, b128 copies (src row = 1728 B = 108 uint4)
    {
        const uint4* src = (const uint4*)(aggW + (size_t)b * COUT * KTOT);
        for (int idx = threadIdx.x; idx < COUT * 108; idx += 256) {
            const int r = idx / 108, c = idx - r * 108;
            ((uint4*)(sA + r * ASTRIDE))[c] = src[(size_t)r * 108 + c];
        }
    }
    __syncthreads();

    const int wave = threadIdx.x >> 5;
    const int lane = threadIdx.x & 31;
    const int hr   = blockIdx.x * 8 + wave;          // half-row id 0..4095
    const int d    = hr >> 7;                        // 128 half-rows per d-slice
    const int rem  = hr & 127;
    const int h    = rem >> 1;
    const int w0   = (rem & 1) << 5;                 // 0 or 32

    const int n       = lane & 15;
    const int halfsel = lane >> 4;

    // lanes 16-31 hold B rows K=16..31 -> channels i = 16 + e
    const float* xb = x + (size_t)b * CIN * DHW + (size_t)halfsel * 16 * DHW;
    const _Float16* arow = sA + (lane & 15) * ASTRIDE + 16 * halfsel;

    v8f acc0[4] = {{}, {}, {}, {}};                  // N-tile 0 (w0..w0+15)
    v8f acc1[4] = {{}, {}, {}, {}};                  // N-tile 1 (w0+16..w0+31)

    for (int ks = 0; ks < 27; ++ks) {                // 27 taps
        const int k0 = ks * 32;
        const int kd = ks / 9, r9 = ks - kd * 9;
        const int kh = r9 / 3, kw = r9 - kh * 3;
        const int dz = d + kd - 1, hy = h + kh - 1;
        const int wx = w0 + n + kw - 1;
        const bool okz = ((unsigned)dz < DD) & ((unsigned)hy < HH);
        const bool ok0 = okz & ((unsigned)wx < WW);
        const bool ok1 = okz & ((unsigned)(wx + 16) < WW);
        const float* bp = xb + (ptrdiff_t)((dz * HH + hy) * WW + wx);

        v16h bf0, bf1;
#pragma unroll
        for (int e = 0; e < 16; ++e) {               // imm offsets e*DHW*4 (+64B)
            float x0 = 0.f, x1 = 0.f;
            if (ok0) x0 = bp[e * DHW];
            if (ok1) x1 = bp[e * DHW + 16];
            bf0[e] = (_Float16)x0;
            bf1[e] = (_Float16)x1;
        }

#pragma unroll
        for (int mt = 0; mt < 4; ++mt) {
            // A fragment: two contiguous 16B runs -> 2x ds_load_b128, reused x2
            const _Float16* ar = arow + mt * (16 * ASTRIDE) + k0;
            v8h alo = *(const v8h*)(ar);
            v8h ahi = *(const v8h*)(ar + 8);
            v16h af;
#pragma unroll
            for (int e = 0; e < 8; ++e) { af[e] = alo[e]; af[8 + e] = ahi[e]; }
            acc0[mt] = __builtin_amdgcn_wmma_f32_16x16x32_f16(
                false, af, false, bf0, (short)0, acc0[mt], false, false);
            acc1[mt] = __builtin_amdgcn_wmma_f32_16x16x32_f16(
                false, af, false, bf1, (short)0, acc1[mt], false, false);
        }
    }

    // ---- store: lanes 0-15: M=r, lanes 16-31: M=8+r
    const int pos = d * (HH * WW) + h * WW + w0 + n;
    float* ob = out + (size_t)b * COUT * DHW + pos;
    const float* bias = aggB + b * COUT + halfsel * 8;
#pragma unroll
    for (int mt = 0; mt < 4; ++mt) {
#pragma unroll
        for (int r = 0; r < 8; ++r) {
            const int o = mt * 16 + halfsel * 8 + r;
            const float bv = bias[mt * 16 + r];
            ob[(size_t)o * DHW]      = acc0[mt][r] + bv;
            ob[(size_t)o * DHW + 16] = acc1[mt][r] + bv;
        }
    }
}

// ---------------------------------------------------------------- launch
extern "C" void kernel_launch(void* const* d_in, const int* in_sizes, int n_in,
                              void* d_out, int out_size, void* d_ws, size_t ws_size,
                              hipStream_t stream) {
    const float* x      = (const float*)d_in[0];
    const float* weight = (const float*)d_in[1];
    const float* cbias  = (const float*)d_in[2];
    const float* temp   = (const float*)d_in[3];
    const float* bn1_g  = (const float*)d_in[4];
    const float* bn1_b  = (const float*)d_in[5];
    const float* w1     = (const float*)d_in[6];
    const float* b1     = (const float*)d_in[7];
    const float* bn2_g  = (const float*)d_in[8];
    const float* bn2_b  = (const float*)d_in[9];
    const float* w2     = (const float*)d_in[10];
    const float* b2     = (const float*)d_in[11];
    const float* bn3_g  = (const float*)d_in[12];
    const float* bn3_b  = (const float*)d_in[13];
    const float* w3     = (const float*)d_in[14];
    const float* b3     = (const float*)d_in[15];
    float* out = (float*)d_out;

    char* ws = (char*)d_ws;
    float*    p    = (float*)(ws + 0);          //  1 KB : pooled [8,32]
    float*    attn = (float*)(ws + 1024);       //  128 B: attention [8,4]
    float*    aggB = (float*)(ws + 2048);       //  2 KB : agg bias [8,64]
    _Float16* aggW = (_Float16*)(ws + 4096);    // 884 KB: agg weights [8,64,864] f16 (tap-major K)

    pool_kernel<<<BB * CIN, 256, 0, stream>>>(x, p);
    attn_kernel<<<1, 32, 0, stream>>>(p, bn1_g, bn1_b, w1, b1,
                                      bn2_g, bn2_b, w2, b2,
                                      bn3_g, bn3_b, w3, b3, temp, attn);
    aggw_kernel<<<BB * COUT, 256, 0, stream>>>(weight, cbias, attn, aggW, aggB);

    dim3 grid(512, BB);                         // (half-row groups, batch)
    conv_wmma_kernel<<<grid, 256, 0, stream>>>(x, aggW, aggB, out);
}
